// LIF_RRF1dplus_Net_21534966022879
// MI455X (gfx1250) — matrix operand order repaired and checked by reading, not compiled, not run on hardware
//
#include <hip/hip_runtime.h>
#include <hip/hip_bf16.h>

typedef __attribute__((ext_vector_type(8)))  _Float16 v8h;
typedef __attribute__((ext_vector_type(16))) _Float16 v16h;
typedef __attribute__((ext_vector_type(8)))  float    v8f;

#define B_DIM   128
#define T_DIM   500
#define F_DIM   34
#define TWIN    5
#define KF      7
#define SF      3
#define L_DIM   10
#define C_OUT   64
#define H_DIM   (L_DIM * C_OUT)          // 640
#define KPAD    96                       // 70 padded to 3x32
#define NROWS   (T_DIM * B_DIM)          // 64000
#define AS_STRIDE 104                    // LDS row stride (halves), conflict-free
#define NSLOTS  64                       // spread slots for BN partial sums

// ---------------------------------------------------------------------------
// Kernel 1: on/off Sobel-difference tensor, f16, layout [b][i][f][tau]
// (8.7 MB -> stays resident in the 192 MB L2 for the GEMM's im2col gathers)
// ---------------------------------------------------------------------------
__global__ __launch_bounds__(256)
void sobel_onoff_kernel(const float* __restrict__ x,
                        const float* __restrict__ w_sobel,
                        _Float16* __restrict__ onoff)
{
    const int idx = blockIdx.x * 256 + threadIdx.x;
    const int total = B_DIM * 2 * F_DIM * T_DIM;
    if (idx >= total) return;
    const float w  = w_sobel[0];
    const int tau  = idx % T_DIM;
    const int f    = (idx / T_DIM) % F_DIM;
    const int i    = (idx / (T_DIM * F_DIM)) % 2;
    const int b    = idx / (T_DIM * F_DIM * 2);
    const float xc = x[(b * F_DIM + f) * T_DIM + tau];
    const float xp = (tau > 0) ? x[(b * F_DIM + f) * T_DIM + tau - 1] : 0.f;
    const float d  = xc - w * xp;
    const float vv = (i == 0) ? fmaxf(d, 0.f) : fmaxf(-d, 0.f);
    onoff[idx] = (_Float16)vv;
}

// ---------------------------------------------------------------------------
// Kernel 2: pack rrf_w into Bt[l][n][kpad] (f16, column-contiguous-K for WMMA
// B fragments), and zero the spread BN-stat accumulators (64 slots x 128).
// ---------------------------------------------------------------------------
__global__ __launch_bounds__(256)
void pack_weights_kernel(const float* __restrict__ rrf_w,
                         _Float16* __restrict__ Bt,
                         float* __restrict__ stats)   // NSLOTS*128 floats
{
    const int idx = blockIdx.x * 256 + threadIdx.x;
    if (idx < NSLOTS * 128) stats[idx] = 0.f;
    if (idx >= L_DIM * C_OUT * KPAD) return;
    const int kk = idx % KPAD;
    const int n  = (idx / KPAD) % C_OUT;
    const int l  = idx / (KPAD * C_OUT);
    _Float16 v = (_Float16)0.f;
    if (kk < 2 * KF * TWIN) {
        const int i  = kk / (KF * TWIN);
        const int rm = kk % (KF * TWIN);
        const int k  = rm / TWIN;
        const int w  = rm % TWIN;
        // rrf_w layout: (C_OUT, L, 2, K, TWIN)
        v = (_Float16)rrf_w[(((n * L_DIM + l) * 2 + i) * KF + k) * TWIN + w];
    }
    Bt[idx] = v;
}

// ---------------------------------------------------------------------------
// Kernel 3: 10 locally-connected GEMMs via v_wmma_f32_16x16x32_f16.
// Block = 128 threads = 4 waves: one 16-row (t,b) tile x all 64 channels for
// one l. A tile gathered from L2-resident onoff into LDS. Epilogue adds bias,
// accumulates per-channel BN sum/sumsq (f32, pre-rounding) into spread slots,
// and stores z as f16: z[(t*B+b)*640 + n*10 + l].
// ---------------------------------------------------------------------------
__global__ __launch_bounds__(128)
void rrf_gemm_wmma_kernel(const _Float16* __restrict__ onoff,
                          const _Float16* __restrict__ Bt,
                          const float* __restrict__ rrf_b,
                          _Float16* __restrict__ z,
                          float* __restrict__ stats)
{
    __shared__ _Float16 As[16 * AS_STRIDE];

    const int l    = blockIdx.y;
    const int t    = blockIdx.x >> 3;           // 8 blocks per t (128/16)
    const int b0   = (blockIdx.x & 7) << 4;
    const int tid  = threadIdx.x;
    const int lane = tid & 31;
    const int wave = tid >> 5;

    // Cooperative gather of the 16x96 f16 A tile (im2col on the fly).
    for (int e = tid; e < 16 * KPAD; e += 128) {
        const int row = e / KPAD;               // -> b = b0 + row
        const int kk  = e % KPAD;
        _Float16 v = (_Float16)0.f;
        if (kk < 2 * KF * TWIN) {
            const int i   = kk / (KF * TWIN);
            const int rm  = kk % (KF * TWIN);
            const int k   = rm / TWIN;
            const int w   = rm % TWIN;
            const int tau = t + w - (TWIN - 1);
            if (tau >= 0) {
                const int f = l * SF + k;
                v = onoff[(((b0 + row) * 2 + i) * F_DIM + f) * T_DIM + tau];
            }
        }
        As[row * AS_STRIDE + kk] = v;
    }
    __syncthreads();

    const int n0 = wave << 4;                   // 16-channel slab per wave
    const int lr = lane & 15;
    const int hi = lane >> 4;

    // B fragment source: Bt[l][n0+lr][k0 + hi*8 + {0..7,16..23}]
    const _Float16* btile = Bt + ((l * C_OUT + n0 + lr) * KPAD) + hi * 8;
    // A fragment source (ISA 16-bit A layout): row lr, halves k0+hi*8+{0..7,16..23}
    const _Float16* arow  = As + lr * AS_STRIDE + hi * 8;

    v8f acc = {};
    #pragma unroll
    for (int c = 0; c < 3; ++c) {
        const int k0 = c * 32;
        const v8h alo = *(const v8h*)(arow  + k0);
        const v8h ahi = *(const v8h*)(arow  + k0 + 16);
        const v8h blo = *(const v8h*)(btile + k0);
        const v8h bhi = *(const v8h*)(btile + k0 + 16);
        v16h a, bb;
        #pragma unroll
        for (int j = 0; j < 8; ++j) {
            a[j]  = alo[j];  a[j + 8]  = ahi[j];
            bb[j] = blo[j];  bb[j + 8] = bhi[j];
        }
        acc = __builtin_amdgcn_wmma_f32_16x16x32_f16(false, a, false, bb,
                                                     (short)0, acc, false, false);
    }

    // Epilogue: bias, f16 store, fused BN partial stats (f32 accumulators).
    const int   n    = n0 + lr;                 // output channel
    const float bias = rrf_b[n * L_DIM + l];
    const long  r0   = (long)t * B_DIM + b0;
    float s = 0.f, q = 0.f;
    #pragma unroll
    for (int v = 0; v < 8; ++v) {
        const float val = acc[v] + bias;
        s += val;
        q += val * val;
        const long r = r0 + v + hi * 8;
        z[r * H_DIM + n * L_DIM + l] = (_Float16)val;
    }
    // Fold half-wave partner (same channel n, rows 8..15) onto lanes 0-15.
    s += __shfl_down(s, 16, 32);
    q += __shfl_down(q, 16, 32);
    if (hi == 0) {
        float* slot = stats + (blockIdx.x & (NSLOTS - 1)) * 128;
        atomicAdd(&slot[n], s);
        atomicAdd(&slot[64 + n], q);
    }
}

// ---------------------------------------------------------------------------
// Kernel 4: fold the spread slots and finalize BN affine params:
// scale = gamma*rsqrt(var+eps), shift = beta - mu*scale.
// ---------------------------------------------------------------------------
__global__ void bn_params_kernel(const float* __restrict__ stats,
                                 const float* __restrict__ gamma,
                                 const float* __restrict__ beta,
                                 float* __restrict__ affine)  // [0..63] scale, [64..127] shift
{
    const int m = threadIdx.x;
    if (m >= C_OUT) return;
    float s = 0.f, q = 0.f;
    for (int sl = 0; sl < NSLOTS; ++sl) {
        s += stats[sl * 128 + m];
        q += stats[sl * 128 + 64 + m];
    }
    const float N   = (float)((long)NROWS * L_DIM);   // 640000
    const float mu  = s / N;
    const float var = q / N - mu * mu;
    const float sc  = gamma[m] * rsqrtf(var + 1e-5f);
    affine[m]      = sc;
    affine[64 + m] = beta[m] - mu * sc;
}

// ---------------------------------------------------------------------------
// Kernel 5: fused BN-apply + PLIF scan + FC. One block per batch element;
// 640 threads = one (m,l) sequence each; coalesced f16 z reads per t; spike
// dot product reduced with wave32 shuffles + tiny LDS pass.
// ---------------------------------------------------------------------------
__global__ __launch_bounds__(H_DIM)
void lif_scan_fc_kernel(const _Float16* __restrict__ z,
                        const float* __restrict__ affine,
                        const float* __restrict__ plif_w,
                        const float* __restrict__ fc_w,
                        const float* __restrict__ fc_b,
                        float* __restrict__ out)
{
    const int b    = blockIdx.x;
    const int h    = threadIdx.x;           // h = m*10 + l
    const int m    = h / L_DIM;
    const int lane = h & 31;
    const int wid  = h >> 5;                // 20 waves

    const float tau_inv = 1.f / (1.f + __expf(-plif_w[0]));
    const float sc  = affine[m];
    const float sh  = affine[64 + m];
    const float wfc = fc_w[h];
    const float bfc = fc_b[0];

    __shared__ float wsum[H_DIM / 32];

    float v = 0.f;
    for (int t = 0; t < T_DIM; ++t) {
        const float zz = (float)z[((long)t * B_DIM + b) * H_DIM + h];
        const float zb = sc * zz + sh;
        v += (zb - v) * tau_inv;
        const float s = (v >= 0.5f) ? 1.f : 0.f;
        v *= (1.f - s);
        float c = s * wfc;
        #pragma unroll
        for (int d = 16; d > 0; d >>= 1) c += __shfl_down(c, d, 32);
        if (lane == 0) wsum[wid] = c;
        __syncthreads();
        if (h == 0) {
            float r = bfc;
            #pragma unroll
            for (int w = 0; w < H_DIM / 32; ++w) r += wsum[w];
            out[b * T_DIM + t] = r;
        }
        __syncthreads();
    }
}

// ---------------------------------------------------------------------------
extern "C" void kernel_launch(void* const* d_in, const int* in_sizes, int n_in,
                              void* d_out, int out_size, void* d_ws, size_t ws_size,
                              hipStream_t stream) {
    const float* x        = (const float*)d_in[0];
    const float* w_sobel  = (const float*)d_in[1];
    const float* rrf_w    = (const float*)d_in[2];
    const float* rrf_b    = (const float*)d_in[3];
    const float* bn_gamma = (const float*)d_in[4];
    const float* bn_beta  = (const float*)d_in[5];
    const float* plif_w   = (const float*)d_in[6];
    const float* fc_w     = (const float*)d_in[7];
    const float* fc_b     = (const float*)d_in[8];
    float* out = (float*)d_out;

    // Workspace layout (offsets 256B-aligned); total ~90.8 MB
    char* ws = (char*)d_ws;
    const size_t onoff_bytes = (size_t)B_DIM * 2 * F_DIM * T_DIM * sizeof(_Float16); //  8,704,000
    const size_t bt_bytes    = (size_t)L_DIM * C_OUT * KPAD * sizeof(_Float16);      //    122,880
    const size_t z_bytes     = (size_t)NROWS * H_DIM * sizeof(_Float16);             // 81,920,000
    const size_t st_bytes    = (size_t)NSLOTS * 128 * sizeof(float);                 //     32,768
    _Float16* onoff  = (_Float16*)(ws);
    _Float16* Bt     = (_Float16*)(ws + onoff_bytes);
    _Float16* z      = (_Float16*)(ws + onoff_bytes + bt_bytes);
    float*    stats  = (float*)   (ws + onoff_bytes + bt_bytes + z_bytes);
    float*    affine = (float*)   (ws + onoff_bytes + bt_bytes + z_bytes + st_bytes);

    // 1) on/off tensor
    {
        const int total = B_DIM * 2 * F_DIM * T_DIM;
        sobel_onoff_kernel<<<(total + 255) / 256, 256, 0, stream>>>(x, w_sobel, onoff);
    }
    // 2) pack weights + zero spread stats
    {
        const int total = L_DIM * C_OUT * KPAD;
        pack_weights_kernel<<<(total + 255) / 256, 256, 0, stream>>>(rrf_w, Bt, stats);
    }
    // 3) 10 x (64000 x 64 x 70) GEMMs on WMMA, fused bias + BN partial stats
    {
        dim3 grid(NROWS / 16, L_DIM);
        rrf_gemm_wmma_kernel<<<grid, 128, 0, stream>>>(onoff, Bt, rrf_b, z, stats);
    }
    // 4) fold stats -> BN affine params
    bn_params_kernel<<<1, 64, 0, stream>>>(stats, bn_gamma, bn_beta, affine);
    // 5) fused BN + LIF scan + FC
    lif_scan_fc_kernel<<<B_DIM, H_DIM, 0, stream>>>(z, affine, plif_w, fc_w, fc_b, out);
}
